// AttentionHead_23398981829147
// MI455X (gfx1250) — matrix operand order repaired and checked by reading, not compiled, not run on hardware
//
#include <hip/hip_runtime.h>

// ---------------------------------------------------------------------------
// Single-head causal attention for MI455X (gfx1250, wave32, WMMA bf16).
//   B=4, S=2048, D_MODEL=1024, D_KEY=128
// HBM-bound problem (~100MB fp32 inputs vs ~15 GFLOP) -> read inputs once,
// convert to bf16 inline, flash-attention (no SxS materialization, no mask
// tensor read), all matmuls on v_wmma_f32_16x16x32_bf16.
// Weights are pre-converted to bf16 ONCE (kernel 0) so the projection GEMM
// does zero conversion work on its B operands and reads half the bytes.
// ---------------------------------------------------------------------------

typedef __attribute__((ext_vector_type(16))) __bf16 v16bf;
typedef __attribute__((ext_vector_type(8)))  __bf16 v8bf;
typedef __attribute__((ext_vector_type(8)))  float  v8f;

#define WMMA_BF16(a, b, c) \
    __builtin_amdgcn_wmma_f32_16x16x32_bf16(false, (a), false, (b), (short)0, (c), false, false)

constexpr int Bn = 4, Sn = 2048, DM = 1024, DK = 128;

// ===========================================================================
// Kernel 0: one-shot fp32 -> bf16 weight conversion (3 x 128 x 1024).
// Each thread converts 8 elements (b128 load -> 4x v_cvt_pk -> b128 store).
// ===========================================================================
__global__ __launch_bounds__(256)
void wconv_kernel(const float* __restrict__ Wq, const float* __restrict__ Wk,
                  const float* __restrict__ Wv, __bf16* __restrict__ Wb)
{
    const int t = blockIdx.x * blockDim.x + threadIdx.x;   // 0 .. 3*128*1024/8-1
    const int per = DK * DM / 8;                           // 16384 per matrix
    const int which = t / per;
    const int off8  = (t % per) * 8;
    const float* W = (which == 0) ? Wq : (which == 1) ? Wk : Wv;
    const float4* p = (const float4*)(W + off8);
    float4 f0 = p[0], f1 = p[1];
    v8bf o;
    o[0] = (__bf16)f0.x; o[1] = (__bf16)f0.y; o[2] = (__bf16)f0.z; o[3] = (__bf16)f0.w;
    o[4] = (__bf16)f1.x; o[5] = (__bf16)f1.y; o[6] = (__bf16)f1.z; o[7] = (__bf16)f1.w;
    *(v8bf*)(Wb + (size_t)which * DK * DM + off8) = o;
}

// ===========================================================================
// Kernel 1: projection GEMM with inline fp32->bf16 on the A operand only.
//   out[s,k] = sum_d in[s,d] * W[k,d]   (NT gemm, both K-contiguous rows)
// One wave (32 lanes) computes a 16-row x 128-col tile.
// Q,K stored row-major bf16 [B][S][128]; V stored transposed [B][128][S]
// so the attention kernel's V B-operand loads are contiguous.
// ===========================================================================
__global__ __launch_bounds__(32)
void proj_kernel(const float* __restrict__ qin, const float* __restrict__ kin,
                 const float* __restrict__ vin, const __bf16* __restrict__ Wb,
                 __bf16* __restrict__ Qb, __bf16* __restrict__ Kb,
                 __bf16* __restrict__ Vt)
{
    const int lane = threadIdx.x & 31;
    const int nn   = lane & 15;   // A: row M  | B: col N | C: col N
    const int hi   = lane >> 4;   // selects K sub-range / M half
    const int s_base = blockIdx.x * 16;
    const int b      = blockIdx.y;
    const int which  = blockIdx.z;        // 0=Q 1=K 2=V

    const float*  src = (which == 0) ? qin : (which == 1) ? kin : vin;
    const __bf16* W   = Wb + (size_t)which * DK * DM;     // bf16, row-major [128][1024]
    src += (size_t)b * Sn * DM;

    v8f acc[8];
#pragma unroll
    for (int n = 0; n < 8; ++n)
#pragma unroll
        for (int i = 0; i < 8; ++i) acc[n][i] = 0.0f;

    const float* arow = src + (size_t)(s_base + nn) * DM;

    for (int kc = 0; kc < DM / 32; ++kc) {
        // ---- A operand (16x32 bf16): a[g*8+i] = arow[kc*32 + g*16 + hi*8 + i]
        v16bf a;
#pragma unroll
        for (int g = 0; g < 2; ++g) {
            const float4* p = (const float4*)(arow + kc * 32 + g * 16 + hi * 8);
            float4 f0 = p[0], f1 = p[1];
            a[g*8+0] = (__bf16)f0.x; a[g*8+1] = (__bf16)f0.y;
            a[g*8+2] = (__bf16)f0.z; a[g*8+3] = (__bf16)f0.w;
            a[g*8+4] = (__bf16)f1.x; a[g*8+5] = (__bf16)f1.y;
            a[g*8+6] = (__bf16)f1.z; a[g*8+7] = (__bf16)f1.w;
        }
        // ---- 8 output N-tiles, B operand (32x16): b[g*8+i] = W[n16+nn][kc*32 + g*16 + hi*8 + i]
#pragma unroll
        for (int n = 0; n < 8; ++n) {
            const __bf16* wrow = W + (size_t)(n * 16 + nn) * DM + kc * 32 + hi * 8;
            v16bf bb;
            v8bf lo = *(const v8bf*)(wrow);
            v8bf hh = *(const v8bf*)(wrow + 16);
#pragma unroll
            for (int i = 0; i < 8; ++i) { bb[i] = lo[i]; bb[8 + i] = hh[i]; }
            acc[n] = WMMA_BF16(a, bb, acc[n]);
        }
    }

    // ---- store: C layout lane=(N=nn), VGPR r -> row M = r + 8*hi
    if (which == 2) {
        // Vt[b][d][s]: lane's d fixed, 8 consecutive s -> one b128 store
#pragma unroll
        for (int n = 0; n < 8; ++n) {
            v8bf o;
#pragma unroll
            for (int r = 0; r < 8; ++r) o[r] = (__bf16)acc[n][r];
            __bf16* dst = Vt + ((size_t)b * DK + (n * 16 + nn)) * Sn + s_base + hi * 8;
            *(v8bf*)dst = o;
        }
    } else {
        __bf16* dst = ((which == 0) ? Qb : Kb) + ((size_t)b * Sn + s_base) * DK;
#pragma unroll
        for (int n = 0; n < 8; ++n)
#pragma unroll
            for (int r = 0; r < 8; ++r)
                dst[(size_t)(r + 8 * hi) * DK + n * 16 + nn] = (__bf16)acc[n][r];
    }
}

// ===========================================================================
// Kernel 2: flash attention, one wave per 16-query tile, causal.
// exp2-domain online softmax; P transposed C-layout -> A-layout through LDS.
// bf16 Q/K/Vt working set = 6MB -> fully L2-resident (192MB) across re-reads.
// ===========================================================================
__global__ __launch_bounds__(32)
void attn_kernel(const __bf16* __restrict__ Qb, const __bf16* __restrict__ Kb,
                 const __bf16* __restrict__ Vt, float* __restrict__ out)
{
    __shared__ __bf16 Pt[16 * 32];   // 1 KB P-tile staging

    const int lane = threadIdx.x & 31;
    const int nn   = lane & 15;
    const int hi   = lane >> 4;
    const int q_base = blockIdx.x * 16;
    const int b      = blockIdx.y;

    const __bf16* Qp = Qb + (size_t)b * Sn * DK;
    const __bf16* Kp = Kb + (size_t)b * Sn * DK;
    const __bf16* Vp = Vt + (size_t)b * DK * Sn;

    // Q A-operand (4 chunks of K=32 over d_key=128), loaded once
    v16bf qa[4];
    const __bf16* qrow = Qp + (size_t)(q_base + nn) * DK;
#pragma unroll
    for (int c = 0; c < 4; ++c) {
        v8bf lo = *(const v8bf*)(qrow + c * 32 + hi * 8);
        v8bf hh = *(const v8bf*)(qrow + c * 32 + 16 + hi * 8);
#pragma unroll
        for (int i = 0; i < 8; ++i) { qa[c][i] = lo[i]; qa[c][8 + i] = hh[i]; }
    }

    const float sm_scale = 0.08838834764831845f * 1.4426950408889634f; // 1/sqrt(128)*log2(e)

    v8f oacc[8];
#pragma unroll
    for (int n = 0; n < 8; ++n)
#pragma unroll
        for (int i = 0; i < 8; ++i) oacc[n][i] = 0.0f;

    float m_i[8], l_i[8];
#pragma unroll
    for (int r = 0; r < 8; ++r) { m_i[r] = -3.0e38f; l_i[r] = 0.0f; }

    for (int j = 0; j <= q_base; j += 32) {
        if (j + 32 <= q_base)   // prefetch next key tile (global_prefetch_b8)
            __builtin_prefetch(Kp + (size_t)(j + 32 + nn) * DK, 0, 1);

        // ---- S = Q K^T for 2 sub-tiles of 16 keys
        v8f sv[2];
#pragma unroll
        for (int jt = 0; jt < 2; ++jt) {
            v8f s;
#pragma unroll
            for (int i = 0; i < 8; ++i) s[i] = 0.0f;
            const __bf16* krow = Kp + (size_t)(j + jt * 16 + nn) * DK;
#pragma unroll
            for (int c = 0; c < 4; ++c) {
                v16bf kb;
                v8bf lo = *(const v8bf*)(krow + c * 32 + hi * 8);
                v8bf hh = *(const v8bf*)(krow + c * 32 + 16 + hi * 8);
#pragma unroll
                for (int i = 0; i < 8; ++i) { kb[i] = lo[i]; kb[8 + i] = hh[i]; }
                s = WMMA_BF16(qa[c], kb, s);
            }
            const int kj = j + jt * 16 + nn;
#pragma unroll
            for (int r = 0; r < 8; ++r) {
                const int qi = q_base + r + 8 * hi;
                sv[jt][r] = (kj <= qi) ? s[r] * sm_scale : -3.0e38f;
            }
        }

        // ---- online softmax (rows live across 16 lanes of same half)
        v8f p0, p1;
#pragma unroll
        for (int r = 0; r < 8; ++r) {
            float mx = fmaxf(sv[0][r], sv[1][r]);
#pragma unroll
            for (int off = 8; off >= 1; off >>= 1)
                mx = fmaxf(mx, __shfl_xor(mx, off, 16));
            const float mnew = fmaxf(m_i[r], mx);
            const float corr = exp2f(m_i[r] - mnew);
            const float a0 = exp2f(sv[0][r] - mnew);
            const float a1 = exp2f(sv[1][r] - mnew);
            float rs = a0 + a1;
#pragma unroll
            for (int off = 8; off >= 1; off >>= 1)
                rs += __shfl_xor(rs, off, 16);
            l_i[r] = l_i[r] * corr + rs;
            m_i[r] = mnew;
            p0[r] = a0; p1[r] = a1;
#pragma unroll
            for (int n = 0; n < 8; ++n) oacc[n][r] *= corr;
        }

        // ---- C-layout P -> row-major LDS [16 rows][32 keys]
#pragma unroll
        for (int r = 0; r < 8; ++r) {
            const int m = r + 8 * hi;
            Pt[m * 32 + nn]      = (__bf16)p0[r];
            Pt[m * 32 + 16 + nn] = (__bf16)p1[r];
        }
        asm volatile("s_wait_dscnt 0" ::: "memory"); // single wave: LDS in-order

        // ---- reload as A operand (16x32)
        v16bf pa;
        {
            v8bf lo = *(const v8bf*)(&Pt[nn * 32 + hi * 8]);
            v8bf hh = *(const v8bf*)(&Pt[nn * 32 + 16 + hi * 8]);
#pragma unroll
            for (int i = 0; i < 8; ++i) { pa[i] = lo[i]; pa[8 + i] = hh[i]; }
        }

        // ---- O += P * V  (V B-operand contiguous thanks to Vt layout)
#pragma unroll
        for (int n = 0; n < 8; ++n) {
            const __bf16* vrow = Vp + (size_t)(n * 16 + nn) * Sn + j + hi * 8;
            v16bf vb;
            v8bf lo = *(const v8bf*)(vrow);
            v8bf hh = *(const v8bf*)(vrow + 16);
#pragma unroll
            for (int i = 0; i < 8; ++i) { vb[i] = lo[i]; vb[8 + i] = hh[i]; }
            oacc[n] = WMMA_BF16(pa, vb, oacc[n]);
        }
    }

    // ---- epilogue: normalize and store f32 output [B][S][128]
    float inv_l[8];
#pragma unroll
    for (int r = 0; r < 8; ++r) inv_l[r] = 1.0f / l_i[r];
    float* orow = out + ((size_t)b * Sn + q_base) * DK;
#pragma unroll
    for (int n = 0; n < 8; ++n)
#pragma unroll
        for (int r = 0; r < 8; ++r)
            orow[(size_t)(r + 8 * hi) * DK + n * 16 + nn] = oacc[n][r] * inv_l[r];
}

// ===========================================================================
extern "C" void kernel_launch(void* const* d_in, const int* in_sizes, int n_in,
                              void* d_out, int out_size, void* d_ws, size_t ws_size,
                              hipStream_t stream)
{
    const float* qin = (const float*)d_in[0];
    const float* kin = (const float*)d_in[1];
    const float* vin = (const float*)d_in[2];
    // d_in[3] = causal mask tensor: intentionally unread (computed analytically,
    // saves 67 MB of HBM traffic; result is identical).
    const float* Wq = (const float*)d_in[4];
    const float* Wk = (const float*)d_in[5];
    const float* Wv = (const float*)d_in[6];
    float* out = (float*)d_out;

    // workspace: Qb(2MB) | Kb(2MB) | Vt(2MB) | Wb(768KB bf16 weights)
    __bf16* Qb = (__bf16*)d_ws;
    __bf16* Kb = Qb + (size_t)Bn * Sn * DK;
    __bf16* Vt = Kb + (size_t)Bn * Sn * DK;
    __bf16* Wb = Vt + (size_t)Bn * Sn * DK;

    // 0) one-shot weight conversion: 3*128*1024/8 = 49152 threads
    wconv_kernel<<<dim3(3 * DK * DM / 8 / 256), dim3(256), 0, stream>>>(Wq, Wk, Wv, Wb);

    // 1) fused convert+projection GEMM
    dim3 blk(32, 1, 1);
    dim3 pg(Sn / 16, Bn, 3);
    proj_kernel<<<pg, blk, 0, stream>>>(qin, kin, vin, Wb, Qb, Kb, Vt);

    // 2) causal flash attention
    dim3 ag(Sn / 16, Bn, 1);
    attn_kernel<<<ag, blk, 0, stream>>>(Qb, Kb, Vt, out);
}
